// MLPPredictor_36112085025194
// MI455X (gfx1250) — matrix operand order repaired and checked by reading, not compiled, not run on hardware
//
#include <hip/hip_runtime.h>

typedef __attribute__((ext_vector_type(16))) _Float16 v16h;
typedef __attribute__((ext_vector_type(8)))  _Float16 v8h;
typedef __attribute__((ext_vector_type(8)))  float    v8f;
typedef __attribute__((ext_vector_type(4)))  float    v4f;

#define D_FEAT   128
#define KDIM     256      // 2*D_FEAT
#define HIDDEN   256
#define OUT_D    3
#define TILE_E   128      // edges per block (8 waves x M=16)
#define W1_STRIDE 264     // padded half stride (132 dwords -> bank spread; 528B = 33*16B aligned)
#define HE_STRIDE 264

// LDS layout (bytes)
#define OFF_W1 0
#define OFF_HE (OFF_W1 + HIDDEN * W1_STRIDE * 2)   // 135168
#define OFF_B1 (OFF_HE + TILE_E * HE_STRIDE * 2)   // +67584
#define OFF_W2 (OFF_B1 + HIDDEN * 4)
#define OFF_B2 (OFF_W2 + OUT_D * HIDDEN * 4)
#define LDS_BYTES (OFF_B2 + 16)                    // ~206.9 KB < 320 KB/WGP

// CDNA5 has a native V_TANH_F32 TRANS op -> one instruction, no branches.
static __device__ __forceinline__ float fast_tanh(float x) {
#if __has_builtin(__builtin_amdgcn_tanhf)
    return __builtin_amdgcn_tanhf(x);
#elif __has_builtin(__builtin_amdgcn_exp2f) && __has_builtin(__builtin_amdgcn_rcpf)
    // branch-free: tanh(x) = sign(x) * (1 - e) / (1 + e),  e = exp(-2|x|)
    const float e = __builtin_amdgcn_exp2f(-2.8853900817779268f * __builtin_fabsf(x));
    const float r = (1.0f - e) * __builtin_amdgcn_rcpf(1.0f + e);
    return __builtin_copysignf(r, x);
#else
    return tanhf(x);
#endif
}

__global__ void __launch_bounds__(256)
edge_mlp_wmma_kernel(const float* __restrict__ x,
                     const long long* __restrict__ eidx,
                     const float* __restrict__ W1,
                     const float* __restrict__ b1,
                     const float* __restrict__ W2,
                     const float* __restrict__ b2,
                     float* __restrict__ out,
                     int nEdges)
{
    extern __shared__ char smem[];
    _Float16* sW1 = (_Float16*)(smem + OFF_W1);
    _Float16* sHe = (_Float16*)(smem + OFF_HE);   // gathered edge features (f16), reused as sH
    float*    sB1 = (float*)(smem + OFF_B1);
    float*    sW2 = (float*)(smem + OFF_W2);
    float*    sB2 = (float*)(smem + OFF_B2);

    const int t    = threadIdx.x;
    const int lane = t & 31;
    const int wave = t >> 5;
    const long long eBase = (long long)blockIdx.x * TILE_E;

    // ---- cooperative: W1 fp32 -> f16 into LDS (coalesced, 16B packed stores) ----
    for (int i = t * 8; i < HIDDEN * KDIM; i += 256 * 8) {
        v4f w0 = *(const v4f*)(W1 + i);
        v4f w1 = *(const v4f*)(W1 + i + 4);
        const int n = i >> 8;      // W1 row (hidden index)
        const int k = i & 255;     // W1 col (input index), multiple of 8
        v8h h;
        h[0] = (_Float16)w0.x; h[1] = (_Float16)w0.y;
        h[2] = (_Float16)w0.z; h[3] = (_Float16)w0.w;
        h[4] = (_Float16)w1.x; h[5] = (_Float16)w1.y;
        h[6] = (_Float16)w1.z; h[7] = (_Float16)w1.w;
        *(v8h*)(sW1 + n * W1_STRIDE + k) = h;
    }
    if (t < HIDDEN) sB1[t] = b1[t];
    for (int i = t; i < OUT_D * HIDDEN; i += 256) sW2[i] = W2[i];
    if (t < OUT_D) sB2[t] = b2[t];

    // ---- gather: thread t loads one 128-float half-row (sender or receiver) ----
    {
        const int eLoc = t >> 1;
        const int half = t & 1;            // 0 = sender feats (k<128), 1 = receiver
        const long long eg = eBase + eLoc;
        long long row = 0;
        if (eg < (long long)nEdges) row = eidx[(long long)half * nEdges + eg];
        const float* src = x + row * D_FEAT;
        _Float16* dst = sHe + eLoc * HE_STRIDE + half * D_FEAT;
        #pragma unroll 4
        for (int k = 0; k < D_FEAT; k += 8) {
            v4f v0 = *(const v4f*)(src + k);
            v4f v1 = *(const v4f*)(src + k + 4);
            v8h h;
            h[0] = (_Float16)v0.x; h[1] = (_Float16)v0.y;
            h[2] = (_Float16)v0.z; h[3] = (_Float16)v0.w;
            h[4] = (_Float16)v1.x; h[5] = (_Float16)v1.y;
            h[6] = (_Float16)v1.z; h[7] = (_Float16)v1.w;
            *(v8h*)(dst + k) = h;
        }
    }
    __syncthreads();

    // ---- per-wave: preload 8 A fragments (K = 8 chunks of 32) into registers ----
    // ISA 16-bit A 16x32 layout: lane group g: halves 0-7 -> K = kc*32 + g*8 .. +7,
    //                            halves 8-15 -> K = kc*32 + 16 + g*8 .. +7
    union AFrag { v16h v; v8h h[2]; };
    AFrag afrag[8];
    {
        const int m = lane & 15;
        const int g = lane >> 4;
        const _Float16* rowp = sHe + (wave * 16 + m) * HE_STRIDE;
        #pragma unroll
        for (int kc = 0; kc < 8; ++kc) {
            afrag[kc].h[0] = *(const v8h*)(rowp + kc * 32 + g * 8);
            afrag[kc].h[1] = *(const v8h*)(rowp + kc * 32 + 16 + g * 8);
        }
    }
    __syncthreads();   // all A-frags captured; sHe region may now be reused as sH

    // ---- layer 1: 16 N-tiles x 8 K-steps of v_wmma_f32_16x16x32_f16 ----
    _Float16* sH = sHe;
    {
        const int n0 = lane & 15;     // B layout: lane -> N column
        const int g  = lane >> 4;     // halves 0-7 -> K = kc*32 + g*8 + i, 8-15 -> kc*32 + 16 + g*8 + i
        for (int nc = 0; nc < 16; ++nc) {
            v8f acc = {};
            const _Float16* bp = sW1 + (nc * 16 + n0) * W1_STRIDE + g * 8;
            #pragma unroll
            for (int kc = 0; kc < 8; ++kc) {
                union BFrag { v16h v; v8h h[2]; } bf;
                bf.h[0] = *(const v8h*)(bp + kc * 32);
                bf.h[1] = *(const v8h*)(bp + kc * 32 + 16);
                acc = __builtin_amdgcn_wmma_f32_16x16x32_f16(
                        false, afrag[kc].v, false, bf.v,
                        (short)0, acc, false, false);
            }
            // D layout: VGPR r -> M = r + 8*g, lane%16 -> N
            const float bias = sB1[nc * 16 + n0];
            #pragma unroll
            for (int r = 0; r < 8; ++r) {
                const float hval = fast_tanh(acc[r] + bias);
                const int mLoc = wave * 16 + g * 8 + r;
                sH[mLoc * HE_STRIDE + nc * 16 + n0] = (_Float16)hval;
            }
        }
    }
    __syncthreads();

    // ---- layer 2 (tiny, VALU): one thread per edge, 3x256 dot + tanh ----
    if (t < TILE_E) {
        const long long eg = eBase + t;
        if (eg < (long long)nEdges) {
            const _Float16* hrow = sH + t * HE_STRIDE;
            float a0 = sB2[0], a1 = sB2[1], a2 = sB2[2];
            #pragma unroll 4
            for (int k = 0; k < HIDDEN; k += 8) {
                const v8h hv8 = *(const v8h*)(hrow + k);
                #pragma unroll
                for (int u = 0; u < 8; ++u) {
                    const float hv = (float)hv8[u];
                    a0 = fmaf(hv, sW2[0 * HIDDEN + k + u], a0);
                    a1 = fmaf(hv, sW2[1 * HIDDEN + k + u], a1);
                    a2 = fmaf(hv, sW2[2 * HIDDEN + k + u], a2);
                }
            }
            float* o = out + eg * OUT_D;
            o[0] = fast_tanh(a0); o[1] = fast_tanh(a1); o[2] = fast_tanh(a2);
        }
    }
}

extern "C" void kernel_launch(void* const* d_in, const int* in_sizes, int n_in,
                              void* d_out, int out_size, void* d_ws, size_t ws_size,
                              hipStream_t stream) {
    (void)n_in; (void)out_size; (void)d_ws; (void)ws_size;
    const float*     x    = (const float*)d_in[0];
    const long long* eidx = (const long long*)d_in[1];   // int64 edge_index [2, E]
    const float*     W1   = (const float*)d_in[2];
    const float*     b1   = (const float*)d_in[3];
    const float*     W2   = (const float*)d_in[4];
    const float*     b2   = (const float*)d_in[5];
    float* out = (float*)d_out;

    const int nEdges  = in_sizes[1] / 2;                  // 640000
    const int nBlocks = (nEdges + TILE_E - 1) / TILE_E;   // 5000

    edge_mlp_wmma_kernel<<<nBlocks, 256, LDS_BYTES, stream>>>(
        x, eidx, W1, b1, W2, b2, out, nEdges);
}
